// RGN_27668179321466
// MI455X (gfx1250) — matrix operand order repaired and checked by reading, not compiled, not run on hardware
//
#include <hip/hip_runtime.h>
#include <hip/hip_bf16.h>
#include <math.h>

typedef __attribute__((ext_vector_type(16))) __bf16 v16bf;
typedef __attribute__((ext_vector_type(8)))  float  v8f;
typedef __attribute__((ext_vector_type(4)))  int    v4i;

#define LSEQ   256
#define BATCH  64
#define NFEAT  41
#define HID    800
#define GATES  3200          // 4*H
#define NROWS  (LSEQ*BATCH)  // 16384
#define KPAD0  64            // layer0 input dim 42 padded to 64
#define XDIM   1600          // 2*H

union AFrag { unsigned u[8]; v16bf v; };
union BFrag { v4i q[2];      v16bf v; };

__device__ __forceinline__ float sigm_(float x) { return 1.0f / (1.0f + expf(-x)); }

// CDNA5 async global->LDS copy (ASYNCcnt path, bypasses VGPRs).
// VDST operand = wave-relative LDS byte offset (low 32 bits of generic shared ptr,
// shared aperture occupies addr[63:32]).
__device__ __forceinline__ void async_ld_b128(void* lds, const void* gaddr) {
  unsigned l = (unsigned)(unsigned long long)lds;
  asm volatile("global_load_async_to_lds_b128 %0, %1, off"
               :: "v"(l), "v"(gaddr) : "memory");
}
__device__ __forceinline__ void async_wait0() {
#if __has_builtin(__builtin_amdgcn_s_wait_asynccnt)
  __builtin_amdgcn_s_wait_asynccnt(0);
#else
  asm volatile("s_wait_asynccnt 0x0" ::: "memory");
#endif
}

// ---------------------------------------------------------------------------
// small prep kernels
// ---------------------------------------------------------------------------
__global__ void __launch_bounds__(256) k_pack_x0(const float* __restrict__ seq,
                                                 __bf16* __restrict__ X0) {
  int idx = blockIdx.x * 256 + threadIdx.x;
  if (idx >= NROWS * KPAD0) return;
  int row = idx >> 6, col = idx & 63;         // row = t*64 + b
  float v = 0.0f;
  if (col < NFEAT) v = seq[(long)row * NFEAT + col];
  else if (col == NFEAT) v = (float)(row >> 6);   // abs position t
  X0[idx] = (__bf16)v;
}

__global__ void __launch_bounds__(256) k_cvt(const float* __restrict__ src,
                                             __bf16* __restrict__ dst,
                                             int N, int Ks, int Kd) {
  int idx = blockIdx.x * 256 + threadIdx.x;
  if (idx >= N * Kd) return;
  int n = idx / Kd, k = idx - n * Kd;
  dst[idx] = (k < Ks) ? (__bf16)src[(long)n * Ks + k] : (__bf16)0.0f;
}

__global__ void __launch_bounds__(256) k_bias(const float* __restrict__ a,
                                              const float* __restrict__ b,
                                              float* __restrict__ o, int n) {
  int i = blockIdx.x * 256 + threadIdx.x;
  if (i < n) o[i] = a[i] + b[i];
}

// ---------------------------------------------------------------------------
// big input-to-hidden GEMM: out(NROWS x GATES) = X(NROWS x K) * W(GATES x K)^T + bias
// grid: (25 n-tiles of 128, 256 m-tiles of 64), 256 threads (8 waves)
// wave w computes a 64(M) x 16(N) strip with 4 wmma accumulators.
// double-buffered async-to-LDS staging overlaps DMA with WMMA.
// gatepre output (210MB > 192MB L2) is streamed with non-temporal stores.
// ---------------------------------------------------------------------------
__global__ void __launch_bounds__(256)
k_gemm_pre(const __bf16* __restrict__ X, int K,
           const __bf16* __restrict__ W,
           const float* __restrict__ bias,
           float* __restrict__ out)
{
  constexpr int CH = 128, ST = 136;   // chunk elems, padded LDS row stride
  __shared__ __align__(16) __bf16 xs[2][64 * ST];
  const int tid  = threadIdx.x;
  const int wave = tid >> 5, lane = tid & 31;
  const int mtile = blockIdx.y * 64;
  const int n0    = blockIdx.x * 128 + wave * 16;
  const int l15 = lane & 15, hs = lane >> 4;
  const long wbase = (long)(n0 + l15) * K;
  v8f acc[4] = {v8f{}, v8f{}, v8f{}, v8f{}};

  auto stage = [&](int buf, int ck, int kw) {
    const int perRow = kw >> 3;                    // 16B chunks per row
    for (int i = tid; i < 64 * perRow; i += 256) {
      const int r = i / perRow, cq = i - r * perRow;
      async_ld_b128(&xs[buf][r * ST + cq * 8],
                    &X[(long)(mtile + r) * K + ck + cq * 8]);
    }
  };

  const int nch = (K + CH - 1) / CH;
  stage(0, 0, (K < CH) ? K : CH);
  for (int c = 0; c < nch; ++c) {
    const int ck = c * CH;
    const int kw = (K - ck < CH) ? (K - ck) : CH;
    async_wait0();
    __syncthreads();                               // chunk c resident; prev compute done
    if (c + 1 < nch) {
      const int ck2 = ck + CH;
      stage((c + 1) & 1, ck2, (K - ck2 < CH) ? (K - ck2) : CH);  // overlap with WMMA below
    }
    const __bf16* xb = xs[c & 1];
    for (int kk = 0; kk < kw; kk += 32) {
      BFrag bf;   // B 32x16: lanes 0-15 hold K=2j,2j+1; lanes 16-31 hold +16
      const v4i* wp = reinterpret_cast<const v4i*>(&W[wbase + ck + kk + hs * 16]);
      bf.q[0] = wp[0]; bf.q[1] = wp[1];
#pragma unroll
      for (int mb = 0; mb < 4; ++mb) {
        AFrag af;   // A 16x32: row = lane&15, VGPR j holds K=16*hs+2j,+1 (contiguous, same as B)
        const int m = mb * 16 + l15;
#pragma unroll
        for (int j = 0; j < 8; ++j) {
          const int kj = kk + (hs << 4) + (j << 1);
          af.u[j] = *reinterpret_cast<const unsigned*>(&xb[m * ST + kj]);
        }
        acc[mb] = __builtin_amdgcn_wmma_f32_16x16x32_bf16(
            false, af.v, false, bf.v, (short)0, acc[mb], false, false);
      }
    }
  }
  const int col = n0 + l15;
  const float bv = bias[col];
#pragma unroll
  for (int mb = 0; mb < 4; ++mb)
#pragma unroll
    for (int r = 0; r < 8; ++r)
      __builtin_nontemporal_store(acc[mb][r] + bv,
          &out[(long)(mtile + mb * 16 + hs * 8 + r) * GATES + col]);
}

// ---------------------------------------------------------------------------
// one recurrent timestep: gates = h_{t-1} @ Whh^T + gatepre; fuse nonlinearity
// grid: 25 blocks (32 hidden units each), 256 threads.
// wave w: gate = w/2, 16-unit column strip; full batch M=64 via 4 accumulators.
// double-buffered async staging (64-elem K chunks) hides DMA on critical path.
// ---------------------------------------------------------------------------
__global__ void __launch_bounds__(256)
k_lstm_step(const __bf16* __restrict__ hprev, int hstride,
            const __bf16* __restrict__ whh,
            const float* __restrict__ gp,     // gatepre rows for this t (B x GATES)
            float* __restrict__ cbuf,         // B x H
            __bf16* __restrict__ hout, int ostride,
            int first)
{
  constexpr int CH = 64, ST = 72;     // chunk elems, padded LDS row stride
  __shared__ __align__(16) __bf16 xs[2][64 * ST];
  __shared__ float gbuf[4 * 64 * 32];           // [gate][batch][unit]
  const int tid  = threadIdx.x;
  const int wave = tid >> 5, lane = tid & 31;
  const int u0   = blockIdx.x * 32;
  const int gate = wave >> 1;
  const int l15 = lane & 15, hs = lane >> 4;
  const int ncol0 = gate * HID + u0 + (wave & 1) * 16;
  const long wbase = (long)(ncol0 + l15) * HID;
  v8f acc[4] = {v8f{}, v8f{}, v8f{}, v8f{}};

  if (!first) {
    auto stage = [&](int buf, int ck, int kw) {
      const int perRow = kw >> 3;
      for (int i = tid; i < 64 * perRow; i += 256) {
        const int r = i / perRow, cq = i - r * perRow;
        async_ld_b128(&xs[buf][r * ST + cq * 8],
                      &hprev[(long)r * hstride + ck + cq * 8]);
      }
    };
    const int nch = (HID + CH - 1) / CH;   // 13 (12x64 + 32)
    stage(0, 0, CH);
    for (int c = 0; c < nch; ++c) {
      const int ck = c * CH;
      const int kw = (HID - ck < CH) ? (HID - ck) : CH;
      async_wait0();
      __syncthreads();
      if (c + 1 < nch) {
        const int ck2 = ck + CH;
        stage((c + 1) & 1, ck2, (HID - ck2 < CH) ? (HID - ck2) : CH);
      }
      const __bf16* xb = xs[c & 1];
      for (int kk = 0; kk < kw; kk += 32) {
        BFrag bf;
        const v4i* wp = reinterpret_cast<const v4i*>(&whh[wbase + ck + kk + hs * 16]);
        bf.q[0] = wp[0]; bf.q[1] = wp[1];
#pragma unroll
        for (int mb = 0; mb < 4; ++mb) {
          AFrag af;
          const int m = mb * 16 + l15;
#pragma unroll
          for (int j = 0; j < 8; ++j) {
            const int kj = kk + (hs << 4) + (j << 1);
            af.u[j] = *reinterpret_cast<const unsigned*>(&xb[m * ST + kj]);
          }
          acc[mb] = __builtin_amdgcn_wmma_f32_16x16x32_bf16(
              false, af.v, false, bf.v, (short)0, acc[mb], false, false);
        }
      }
    }
  }
  // scatter accumulators to LDS: gbuf[gate][m][ucol]  (disjoint per wave)
  {
    const int col = (wave & 1) * 16 + l15;
#pragma unroll
    for (int mb = 0; mb < 4; ++mb)
#pragma unroll
      for (int r = 0; r < 8; ++r) {
        const int m = mb * 16 + hs * 8 + r;
        gbuf[(gate * 64 + m) * 32 + col] = acc[mb][r];
      }
  }
  __syncthreads();
  // gate fusion: 64 batch x 32 units; gatepre is a one-shot 210MB stream -> NT loads
  for (int idx = tid; idx < 64 * 32; idx += 256) {
    const int u = idx & 31, b = idx >> 5;
    const int ug = u0 + u;
    const float* gpr = gp + (long)b * GATES;
    const float gi = gbuf[(0 * 64 + b) * 32 + u] + __builtin_nontemporal_load(&gpr[ug]);
    const float gf = gbuf[(1 * 64 + b) * 32 + u] + __builtin_nontemporal_load(&gpr[HID + ug]);
    const float gg = gbuf[(2 * 64 + b) * 32 + u] + __builtin_nontemporal_load(&gpr[2 * HID + ug]);
    const float go = gbuf[(3 * 64 + b) * 32 + u] + __builtin_nontemporal_load(&gpr[3 * HID + ug]);
    const float cold = first ? 0.0f : cbuf[(long)b * HID + ug];
    const float cn = sigm_(gf) * cold + sigm_(gi) * tanhf(gg);
    const float h  = sigm_(go) * tanhf(cn);
    cbuf[(long)b * HID + ug] = cn;
    hout[(long)b * ostride + ug] = (__bf16)h;
  }
}

// ---------------------------------------------------------------------------
// head: logits -> softmax -> atan2 angles. one wave per (t,b) row.
// ---------------------------------------------------------------------------
__global__ void __launch_bounds__(256)
k_head(const __bf16* __restrict__ X2, const float* __restrict__ LW,
       const float* __restrict__ LB, const float* __restrict__ AL,
       float* __restrict__ ang)
{
  const int wave = threadIdx.x >> 5, lane = threadIdx.x & 31;
  const int row = blockIdx.x * 8 + wave;
  const __bf16* xr = X2 + (long)row * XDIM;
  float a[20];
#pragma unroll
  for (int u = 0; u < 20; ++u) a[u] = 0.0f;
  for (int k = lane; k < XDIM; k += 32) {
    const float xv = (float)xr[k];
#pragma unroll
    for (int u = 0; u < 20; ++u) a[u] += xv * LW[u * XDIM + k];
  }
#pragma unroll
  for (int u = 0; u < 20; ++u) {
    float v = a[u];
#pragma unroll
    for (int off = 16; off > 0; off >>= 1) v += __shfl_xor(v, off, 32);
    a[u] = v + LB[u];
  }
  if (lane == 0) {
    float mx = a[0];
#pragma unroll
    for (int u = 1; u < 20; ++u) mx = fmaxf(mx, a[u]);
    float s = 0.0f;
#pragma unroll
    for (int u = 0; u < 20; ++u) { a[u] = expf(a[u] - mx); s += a[u]; }
    const float inv = 1.0f / s;
    float sv0 = 0, sv1 = 0, sv2 = 0, cv0 = 0, cv1 = 0, cv2 = 0;
    for (int u = 0; u < 20; ++u) {
      const float p = a[u] * inv;
      const float a0 = AL[u * 3 + 0], a1 = AL[u * 3 + 1], a2 = AL[u * 3 + 2];
      sv0 += p * sinf(a0); cv0 += p * cosf(a0);
      sv1 += p * sinf(a1); cv1 += p * cosf(a1);
      sv2 += p * sinf(a2); cv2 += p * cosf(a2);
    }
    ang[(long)row * 3 + 0] = atan2f(sv0, cv0);
    ang[(long)row * 3 + 1] = atan2f(sv1, cv1);
    ang[(long)row * 3 + 2] = atan2f(sv2, cv2);
  }
}

// ---------------------------------------------------------------------------
// geometric scan: one thread per batch chain, 255 sequential extension steps
// ---------------------------------------------------------------------------
__global__ void k_geom(const float* __restrict__ ang, float* __restrict__ out)
{
  const int b = threadIdx.x;
  if (b >= BATCH) return;
  const float th[3] = {2.034f, 2.119f, 1.937f};
  const float rl[3] = {1.329f, 1.459f, 1.525f};
  float A0 = 0.f, A1 = 0.f, A2 = 1.f;
  float B0 = 0.f, B1 = 1.f, B2 = 0.f;
  float C0 = 1.f, C1 = 0.f, C2 = 0.f;
  out[(0 * BATCH + b) * 3 + 0] = A0; out[(0 * BATCH + b) * 3 + 1] = A1; out[(0 * BATCH + b) * 3 + 2] = A2;
  out[(1 * BATCH + b) * 3 + 0] = B0; out[(1 * BATCH + b) * 3 + 1] = B1; out[(1 * BATCH + b) * 3 + 2] = B2;
  out[(2 * BATCH + b) * 3 + 0] = C0; out[(2 * BATCH + b) * 3 + 1] = C1; out[(2 * BATCH + b) * 3 + 2] = C2;
  for (int t = 1; t < LSEQ; ++t) {
    for (int i = 0; i < 3; ++i) {
      const float P = ang[((long)t * BATCH + b) * 3 + i];
      const float ca = cosf(th[i]), sa = sinf(th[i]);
      const float d0 = -rl[i] * ca;
      const float d1 = rl[i] * cosf(P) * sa;
      const float d2 = rl[i] * sinf(P) * sa;
      float bc0 = C0 - B0, bc1 = C1 - B1, bc2 = C2 - B2;
      float r = rsqrtf(bc0 * bc0 + bc1 * bc1 + bc2 * bc2);
      bc0 *= r; bc1 *= r; bc2 *= r;
      const float ab0 = B0 - A0, ab1 = B1 - A1, ab2 = B2 - A2;
      float n0 = ab1 * bc2 - ab2 * bc1;
      float n1 = ab2 * bc0 - ab0 * bc2;
      float n2 = ab0 * bc1 - ab1 * bc0;
      r = rsqrtf(n0 * n0 + n1 * n1 + n2 * n2);
      n0 *= r; n1 *= r; n2 *= r;
      const float m0 = n1 * bc2 - n2 * bc1;
      const float m1 = n2 * bc0 - n0 * bc2;
      const float m2 = n0 * bc1 - n1 * bc0;
      const float D0 = bc0 * d0 + m0 * d1 + n0 * d2 + C0;
      const float D1 = bc1 * d0 + m1 * d1 + n1 * d2 + C1;
      const float D2 = bc2 * d0 + m2 * d1 + n2 * d2 + C2;
      const long row = 3 + (long)(t - 1) * 3 + i;
      out[(row * BATCH + b) * 3 + 0] = D0;
      out[(row * BATCH + b) * 3 + 1] = D1;
      out[(row * BATCH + b) * 3 + 2] = D2;
      A0 = B0; A1 = B1; A2 = B2;
      B0 = C0; B1 = C1; B2 = C2;
      C0 = D0; C1 = D1; C2 = D2;
    }
  }
}

// ---------------------------------------------------------------------------
// host orchestration
// ---------------------------------------------------------------------------
extern "C" void kernel_launch(void* const* d_in, const int* in_sizes, int n_in,
                              void* d_out, int out_size, void* d_ws, size_t ws_size,
                              hipStream_t stream)
{
  (void)in_sizes; (void)n_in; (void)out_size; (void)ws_size;
  const float* seq = (const float*)d_in[0];
  const float* wih[4]; const float* whhp[4]; const float* bih[4]; const float* bhh[4];
  for (int i = 0; i < 4; ++i) {
    wih[i]  = (const float*)d_in[2 + i];
    whhp[i] = (const float*)d_in[6 + i];
    bih[i]  = (const float*)d_in[10 + i];
    bhh[i]  = (const float*)d_in[14 + i];
  }
  const float* LW = (const float*)d_in[18];
  const float* LB = (const float*)d_in[19];
  const float* AL = (const float*)d_in[20];

  char* ws = (char*)d_ws;
  size_t off = 0;
  auto carve = [&](size_t bytes) -> char* {
    char* p = ws + off;
    off += (bytes + 255) & ~(size_t)255;
    return p;
  };

  __bf16* X0 = (__bf16*)carve((size_t)NROWS * KPAD0 * 2);
  __bf16* wb0[2];
  for (int d = 0; d < 2; ++d) wb0[d] = (__bf16*)carve((size_t)GATES * KPAD0 * 2);
  __bf16* whb[4];
  for (int i = 0; i < 4; ++i) whb[i] = (__bf16*)carve((size_t)GATES * HID * 2);
  __bf16* wb1[2];
  for (int d = 0; d < 2; ++d) wb1[d] = (__bf16*)carve((size_t)GATES * XDIM * 2);
  float* bs[4];
  for (int i = 0; i < 4; ++i) bs[i] = (float*)carve((size_t)GATES * 4);
  float*  gp   = (float*) carve((size_t)NROWS * GATES * 4);   // ~210 MB, reused per pass
  __bf16* X1   = (__bf16*)carve((size_t)NROWS * XDIM * 2);
  __bf16* X2   = (__bf16*)carve((size_t)NROWS * XDIM * 2);
  float*  cbuf = (float*) carve((size_t)BATCH * HID * 4);
  float*  angs = (float*) carve((size_t)NROWS * 3 * 4);

  // weight / input conversion to bf16 (weights stay hot in 192MB L2)
  k_pack_x0<<<(NROWS * KPAD0 + 255) / 256, 256, 0, stream>>>(seq, X0);
  for (int d = 0; d < 2; ++d)
    k_cvt<<<(GATES * KPAD0 + 255) / 256, 256, 0, stream>>>(wih[d], wb0[d], GATES, NFEAT + 1, KPAD0);
  for (int i = 0; i < 4; ++i)
    k_cvt<<<(GATES * HID + 255) / 256, 256, 0, stream>>>(whhp[i], whb[i], GATES, HID, HID);
  for (int d = 0; d < 2; ++d)
    k_cvt<<<(GATES * XDIM + 255) / 256, 256, 0, stream>>>(wih[2 + d], wb1[d], GATES, XDIM, XDIM);
  for (int i = 0; i < 4; ++i)
    k_bias<<<(GATES + 255) / 256, 256, 0, stream>>>(bih[i], bhh[i], bs[i], GATES);

  auto run_dir = [&](const __bf16* Xin, int K, const __bf16* wihB, const float* bv,
                     const __bf16* whhB, __bf16* Xout, int coloff, bool rev) {
    // parallel-over-time input GEMM (peak WMMA)
    k_gemm_pre<<<dim3(25, 256), 256, 0, stream>>>(Xin, K, wihB, bv, gp);
    // sequential recurrence
    for (int s = 0; s < LSEQ; ++s) {
      const int t = rev ? (LSEQ - 1 - s) : s;
      const int tp = rev ? (t + 1) : (t - 1);
      const int first = (s == 0) ? 1 : 0;
      const __bf16* hp = first ? (const __bf16*)nullptr
                               : (Xout + (size_t)tp * BATCH * XDIM + coloff);
      k_lstm_step<<<25, 256, 0, stream>>>(hp, XDIM, whhB,
                                          gp + (size_t)t * BATCH * GATES,
                                          cbuf,
                                          Xout + (size_t)t * BATCH * XDIM + coloff, XDIM,
                                          first);
    }
  };

  run_dir(X0, KPAD0, wb0[0], bs[0], whb[0], X1, 0,   false);  // layer0 fwd
  run_dir(X0, KPAD0, wb0[1], bs[1], whb[1], X1, HID, true);   // layer0 rev
  run_dir(X1, XDIM,  wb1[0], bs[2], whb[2], X2, 0,   false);  // layer1 fwd
  run_dir(X1, XDIM,  wb1[1], bs[3], whb[3], X2, HID, true);   // layer1 rev

  k_head<<<NROWS / 8, 256, 0, stream>>>(X2, LW, LB, AL, angs);
  k_geom<<<1, 64, 0, stream>>>(angs, (float*)d_out);
}
